// MolDecoder_10007273800066
// MI455X (gfx1250) — matrix operand order repaired and checked by reading, not compiled, not run on hardware
//
#include <hip/hip_runtime.h>
#include <math.h>

// ---- problem dims ----
#define BB 64
#define VV 2048
#define ENC 128
#define TT 100
#define EMB 256
#define DEC 512
#define ATT 256
#define VOCAB 29
#define KCAT (EMB + ENC + DEC)   // 896: [emb | gated awe | h]

typedef unsigned short u16;
typedef __attribute__((ext_vector_type(16))) __bf16 v16bf;
typedef __attribute__((ext_vector_type(8)))  float  v8f;

union Frag { v16bf v; unsigned int d[8]; };

__device__ __forceinline__ u16 f2bf(float f) {
  unsigned int u = __builtin_bit_cast(unsigned int, f);
  unsigned int r = (u + 0x7FFFu + ((u >> 16) & 1u)) >> 16;   // RNE
  return (u16)r;
}
__device__ __forceinline__ float bf2f(unsigned int lo16) {
  return __builtin_bit_cast(float, lo16 << 16);
}
__device__ __forceinline__ float sigm(float x) { return 1.f / (1.f + __expf(-x)); }

// ---------------- generic convert / zero ----------------
__global__ void cvt_bf16(const float* __restrict__ in, u16* __restrict__ out, int n) {
  int i = blockIdx.x * 256 + threadIdx.x;
  if (i < n) out[i] = f2bf(in[i]);
}

// fused LSTM weight: Wcat[n][k] = k<384 ? W_ih[n][k] : W_hh[n][k-384]
__global__ void build_wcat(const float* __restrict__ Wih, const float* __restrict__ Whh,
                           u16* __restrict__ wcat) {
  int i = blockIdx.x * 256 + threadIdx.x;            // 2048*896
  if (i >= 4 * DEC * KCAT) return;
  int n = i / KCAT, k = i - n * KCAT;
  float v = (k < EMB + ENC) ? Wih[n * (EMB + ENC) + k] : Whh[n * DEC + (k - (EMB + ENC))];
  wcat[i] = f2bf(v);
}
__global__ void build_bcat(const float* bih, const float* bhh, float* bcat) {
  int i = blockIdx.x * 256 + threadIdx.x;
  if (i < 4 * DEC) bcat[i] = bih[i] + bhh[i];
}
// fused attention-proj weight: rows 0..255 = W_dec (K=512), rows 256..383 = W_beta
__global__ void build_wdg(const float* __restrict__ Wdec, const float* __restrict__ Wbeta,
                          const float* __restrict__ bdec, const float* __restrict__ bbeta,
                          u16* __restrict__ wdg, float* __restrict__ bdg) {
  int i = blockIdx.x * 256 + threadIdx.x;            // 384*512
  if (i < ATT + ENC) bdg[i] = (i < ATT) ? bdec[i] : bbeta[i - ATT];
  if (i >= (ATT + ENC) * DEC) return;
  int n = i / DEC, k = i - n * DEC;
  float v = (n < ATT) ? Wdec[n * DEC + k] : Wbeta[(n - ATT) * DEC + k];
  wdg[i] = f2bf(v);
}

// ---------------- generic WMMA bf16 GEMM ----------------
// C[M,N] = A[M,K](bf16,row-major,lda) * W[N,K]^T (bf16 row-major,ldw) + bias
// act: 0=none 1=sigmoid 2=sigmoid for n>=ATT ; lens!=null: zero rows where t>=lens[m]
__global__ void __launch_bounds__(128)
wmma_gemm(const u16* __restrict__ A, int lda, const u16* __restrict__ W, int ldw,
          const float* __restrict__ bias,
          float* __restrict__ outF, int ldcF, u16* __restrict__ outB, int ldcB,
          int M, int N, int K, int act, const int* __restrict__ lens, int t) {
  int lane = threadIdx.x & 31;
  int wid  = threadIdx.x >> 5;
  int ntN  = (N + 15) >> 4;
  int total = (M >> 4) * ntN;
  int tile = blockIdx.x * 4 + wid;                   // wave-uniform
  if (tile >= total) return;                         // whole-wave exit: EXEC stays full
  int mt = tile / ntN, nt = tile - mt * ntN;
  int m0 = mt << 4, n0 = nt << 4;
  int l15 = lane & 15, khalf = lane >> 4;
  int am = m0 + l15;
  int bn = n0 + l15; if (bn >= N) bn = N - 1;        // clamp ragged N (VOCAB=29)
  const u16* Arow = A + (size_t)am * lda;
  const u16* Wrow = W + (size_t)bn * ldw;
  v8f acc = {};
  for (int k0 = 0; k0 < K; k0 += 32) {
    Frag a, b;
#pragma unroll
    for (int i = 0; i < 8; ++i) {
      // A 16x32 bf16 layout (ISA 7.12.2): vgpr i -> K = 2*(i%4) + 16*(i/4) + 8*khalf
      int ka = k0 + 2 * (i & 3) + ((i >> 2) << 4) + (khalf << 3);
      a.d[i] = *(const unsigned int*)(Arow + ka);
      // B 32x16: same K mapping as A: vgpr i -> K = 2*(i%4) + 16*(i/4) + 8*khalf
      int kb = k0 + 2 * (i & 3) + ((i >> 2) << 4) + (khalf << 3);
      b.d[i] = *(const unsigned int*)(Wrow + kb);
    }
    acc = __builtin_amdgcn_wmma_f32_16x16x32_bf16(false, a.v, false, b.v,
                                                  (short)0, acc, false, false);
  }
  int cn = n0 + l15;
  if (cn < N) {
    float bb = bias ? bias[cn] : 0.f;
#pragma unroll
    for (int r = 0; r < 8; ++r) {
      int cm = m0 + r + (khalf << 3);                // C/D: M = r + 8*(lane>=16)
      float v = acc[r] + bb;
      if (act == 1 || (act == 2 && cn >= ATT)) v = sigm(v);
      if (lens && t >= lens[cm]) v = 0.f;
      if (outF) outF[(size_t)cm * ldcF + cn] = v;
      if (outB) outB[(size_t)cm * ldcB + cn] = f2bf(v);
    }
  }
}

// ---------------- mean over voxels ----------------
__global__ void mean_enc_k(const float* __restrict__ enc, float* __restrict__ mean,
                           u16* __restrict__ mean_bf) {
  int i = blockIdx.x * 128 + threadIdx.x;            // 64*128
  if (i >= BB * ENC) return;
  int b = i >> 7, e = i & 127;
  const float* p = enc + (size_t)b * VV * ENC + e;
  float s = 0.f;
  for (int v = 0; v < VV; ++v) s += p[(size_t)v * ENC];
  s *= (1.f / (float)VV);
  mean[i] = s; mean_bf[i] = f2bf(s);
}

// ============ fused per-step attention mega-kernel (one block per b) ============
// scores -> softmax -> masked alphas out -> awe -> x=[emb | gate*awe | h]
__global__ void __launch_bounds__(1024)
attn_mega(const u16* __restrict__ encatt, const u16* __restrict__ enc_bf,
          const float* __restrict__ dg,           // [B][384] = [dec_att | sigmoid gate]
          const float* __restrict__ w_full, const float* __restrict__ b_full,
          const u16* __restrict__ emb_bf, const int* __restrict__ caps,
          const u16* __restrict__ h_bf, float* __restrict__ alphas_out,
          const int* __restrict__ lens, u16* __restrict__ x_bf, int t) {
  int b = blockIdx.x;
  int tid = threadIdx.x, lane = tid & 31, wid = tid >> 5;
  __shared__ __align__(16) float s_dec[ATT];        // 1 KB
  __shared__ __align__(16) float s_gate[ENC];       // 0.5 KB
  __shared__ __align__(16) float s_wf[ATT];         // 1 KB
  __shared__ __align__(16) float s_sc[VV];          // 8 KB scores -> alpha
  __shared__ __align__(16) float s_awe[8 * ENC];    // 4 KB partial awe
  __shared__ float s_red[32];
  __shared__ float s_b0, s_b1;

  // --- stage operands into LDS via CDNA5 async global->LDS (ASYNCcnt-tracked) ---
  if (tid < 160) {
    const float* g;
    unsigned l;
    if (tid < 64)      { g = dg + (size_t)b * 384 + tid * 4;
                         l = (unsigned)(size_t)(s_dec + tid * 4); }
    else if (tid < 96) { g = dg + (size_t)b * 384 + 256 + (tid - 64) * 4;
                         l = (unsigned)(size_t)(s_gate + (tid - 64) * 4); }
    else               { g = w_full + (tid - 96) * 4;
                         l = (unsigned)(size_t)(s_wf + (tid - 96) * 4); }
    asm volatile("global_load_async_to_lds_b128 %0, %1, off"
                 :: "v"(l), "v"(g) : "memory");
  }
  asm volatile("s_wait_asynccnt 0" ::: "memory");
  __syncthreads();

  // --- scores: each wave owns rows v = wid, wid+32, ... (64 rows/wave) ---
  int a0 = lane * 8;
  for (int v = wid; v < VV; v += 32) {
    const u16* row = encatt + (((size_t)b * VV + v) << 8);   // *ATT
    __builtin_prefetch(row + 32 * ATT, 0, 0);                // global_prefetch
    const unsigned int* rp = (const unsigned int*)(row + a0);
    float acc = 0.f;
#pragma unroll
    for (int j = 0; j < 4; ++j) {
      unsigned int d = rp[j];
      acc += fmaxf(bf2f(d & 0xffffu) + s_dec[a0 + 2 * j],     0.f) * s_wf[a0 + 2 * j];
      acc += fmaxf(bf2f(d >> 16)     + s_dec[a0 + 2 * j + 1], 0.f) * s_wf[a0 + 2 * j + 1];
    }
#pragma unroll
    for (int o = 16; o > 0; o >>= 1) acc += __shfl_xor(acc, o, 32);
    if (lane == 0) s_sc[v] = acc + b_full[0];
  }
  __syncthreads();

  // --- block softmax over 2048 scores (2 per thread) ---
  float a0v = s_sc[tid], a1v = s_sc[tid + 1024];
  float mx = fmaxf(a0v, a1v);
#pragma unroll
  for (int o = 16; o > 0; o >>= 1) mx = fmaxf(mx, __shfl_xor(mx, o, 32));
  if (lane == 0) s_red[wid] = mx;
  __syncthreads();
  if (wid == 0) {
    float m2 = s_red[lane];
#pragma unroll
    for (int o = 16; o > 0; o >>= 1) m2 = fmaxf(m2, __shfl_xor(m2, o, 32));
    if (lane == 0) s_b0 = m2;
  }
  __syncthreads();
  mx = s_b0;
  float e0 = __expf(a0v - mx), e1 = __expf(a1v - mx);
  float sm = e0 + e1;
#pragma unroll
  for (int o = 16; o > 0; o >>= 1) sm += __shfl_xor(sm, o, 32);
  if (lane == 0) s_red[wid] = sm;
  __syncthreads();
  if (wid == 0) {
    float s2 = s_red[lane];
#pragma unroll
    for (int o = 16; o > 0; o >>= 1) s2 += __shfl_xor(s2, o, 32);
    if (lane == 0) s_b1 = 1.f / s2;
  }
  __syncthreads();
  float inv = s_b1;
  float al0 = e0 * inv, al1 = e1 * inv;
  s_sc[tid] = al0; s_sc[tid + 1024] = al1;
  int m = (t < lens[b]);
  float* ao = alphas_out + ((size_t)b * TT + t) * VV;
  ao[tid] = m ? al0 : 0.f;
  ao[tid + 1024] = m ? al1 : 0.f;
  __syncthreads();

  // --- awe[e] = sum_v alpha[v]*enc[b,v,e], 8 v-slices of 256 ---
  {
    int e = tid & 127, q = tid >> 7;
    const u16* eb = enc_bf + ((size_t)b * VV + (size_t)q * 256) * ENC + e;
    __builtin_prefetch(eb, 0, 0);
    float acc = 0.f;
    for (int v = 0; v < 256; ++v) acc += s_sc[q * 256 + v] * bf2f(eb[(size_t)v * ENC]);
    s_awe[q * ENC + e] = acc;
  }
  __syncthreads();

  // --- assemble x = [emb_t | gate*awe | h] in bf16 ---
  u16* xr = x_bf + (size_t)b * KCAT;
  if (tid < ENC) {
    float s = 0.f;
#pragma unroll
    for (int q = 0; q < 8; ++q) s += s_awe[q * ENC + tid];
    xr[EMB + tid] = f2bf(s_gate[tid] * s);
  }
  if (tid < EMB) {
    int tok = caps[b * TT + t];
    xr[tid] = emb_bf[tok * EMB + tid];
  }
  if (tid < DEC) xr[EMB + ENC + tid] = h_bf[(size_t)b * DEC + tid];
}

// ---------------- pointwise LSTM cell update (torch gate order i,f,g,o) ----------------
__global__ void lstm_update(const float* __restrict__ gates, float* __restrict__ h,
                            float* __restrict__ c, u16* __restrict__ h_bf,
                            const int* __restrict__ lens, int t) {
  int i = blockIdx.x * 256 + threadIdx.x;            // 64*512
  if (i >= BB * DEC) return;
  int b = i >> 9, d = i & 511;
  const float* g = gates + (size_t)b * 4 * DEC;
  float ig = sigm(g[d]);
  float fg = sigm(g[DEC + d]);
  float gg = tanhf(g[2 * DEC + d]);
  float og = sigm(g[3 * DEC + d]);
  float cn = fg * c[i] + ig * gg;
  float hn = og * tanhf(cn);
  if (t < lens[b]) { c[i] = cn; h[i] = hn; h_bf[i] = f2bf(hn); }
}

__global__ void lens_out_k(const int* __restrict__ lens, float* __restrict__ out) {
  int b = threadIdx.x;
  if (b < BB) out[b] = (float)lens[b];
}

// =============================================================================
extern "C" void kernel_launch(void* const* d_in, const int* in_sizes, int n_in,
                              void* d_out, int out_size, void* d_ws, size_t ws_size,
                              hipStream_t stream) {
  const float* enc     = (const float*)d_in[0];
  const int*   caps    = (const int*)  d_in[1];
  const int*   lens    = (const int*)  d_in[2];
  const float* emb_W   = (const float*)d_in[3];
  const float* W_enc   = (const float*)d_in[4];
  const float* b_enc   = (const float*)d_in[5];
  const float* W_dec   = (const float*)d_in[6];
  const float* b_dec   = (const float*)d_in[7];
  const float* w_full  = (const float*)d_in[8];
  const float* b_full  = (const float*)d_in[9];
  const float* W_ih    = (const float*)d_in[10];
  const float* b_ih    = (const float*)d_in[11];
  const float* W_hh    = (const float*)d_in[12];
  const float* b_hh    = (const float*)d_in[13];
  const float* W_inh   = (const float*)d_in[14];
  const float* b_inh   = (const float*)d_in[15];
  const float* W_inc   = (const float*)d_in[16];
  const float* b_inc   = (const float*)d_in[17];
  const float* W_beta  = (const float*)d_in[18];
  const float* b_beta  = (const float*)d_in[19];
  const float* W_fin   = (const float*)d_in[20];
  const float* b_fin   = (const float*)d_in[21];

  // ---- workspace carve ----
  char* base = (char*)d_ws;
  size_t off = 0;
  auto alloc = [&](size_t bytes) -> void* {
    off = (off + 255) & ~(size_t)255;
    void* p = base + off;
    off += bytes;
    return p;
  };
  u16* enc_bf    = (u16*)alloc((size_t)BB * VV * ENC * 2);   // 32 MB  (L2-resident)
  u16* encatt_bf = (u16*)alloc((size_t)BB * VV * ATT * 2);   // 64 MB  (L2-resident)
  u16* emb_bf    = (u16*)alloc((size_t)VOCAB * EMB * 2);
  u16* wenc_bf   = (u16*)alloc((size_t)ATT * ENC * 2);
  u16* wdg_bf    = (u16*)alloc((size_t)(ATT + ENC) * DEC * 2);
  float* bdg     = (float*)alloc((size_t)(ATT + ENC) * 4);
  u16* wfin_bf   = (u16*)alloc((size_t)VOCAB * DEC * 2);
  u16* winh_bf   = (u16*)alloc((size_t)DEC * ENC * 2);
  u16* winc_bf   = (u16*)alloc((size_t)DEC * ENC * 2);
  u16* wcat_bf   = (u16*)alloc((size_t)4 * DEC * KCAT * 2);
  float* bcat    = (float*)alloc((size_t)4 * DEC * 4);
  float* meanv   = (float*)alloc((size_t)BB * ENC * 4);
  u16*   mean_bf = (u16*)alloc((size_t)BB * ENC * 2);
  float* h       = (float*)alloc((size_t)BB * DEC * 4);
  float* c       = (float*)alloc((size_t)BB * DEC * 4);
  u16*   h_bf    = (u16*)alloc((size_t)BB * DEC * 2);
  float* dg      = (float*)alloc((size_t)BB * (ATT + ENC) * 4);
  u16*   x_bf    = (u16*)alloc((size_t)BB * KCAT * 2);
  float* gatesb  = (float*)alloc((size_t)BB * 4 * DEC * 4);
  (void)ws_size; (void)n_in; (void)in_sizes; (void)out_size;

  float* outP = (float*)d_out;                       // [B,T,VOCAB]
  float* outA = outP + (size_t)BB * TT * VOCAB;      // [B,T,V]
  float* outL = outA + (size_t)BB * TT * VV;         // [B]

  auto cv = [&](const float* in, u16* out, int n) {
    cvt_bf16<<<(n + 255) / 256, 256, 0, stream>>>(in, out, n);
  };
  auto gemm = [&](const u16* A, int lda, const u16* W, int ldw, const float* bias,
                  float* oF, int ldF, u16* oB, int ldB,
                  int M, int N, int K, int act, const int* L, int t) {
    int tiles = (M >> 4) * ((N + 15) >> 4);
    wmma_gemm<<<(tiles + 3) / 4, 128, 0, stream>>>(A, lda, W, ldw, bias,
                                                   oF, ldF, oB, ldB, M, N, K, act, L, t);
  };

  // ---- prep (loop-invariant) ----
  cv(enc,   enc_bf,   BB * VV * ENC);
  cv(emb_W, emb_bf,   VOCAB * EMB);
  cv(W_enc, wenc_bf,  ATT * ENC);
  cv(W_fin, wfin_bf,  VOCAB * DEC);
  cv(W_inh, winh_bf,  DEC * ENC);
  cv(W_inc, winc_bf,  DEC * ENC);
  build_wdg<<<((ATT + ENC) * DEC + 255) / 256, 256, 0, stream>>>(W_dec, W_beta,
                                                                 b_dec, b_beta, wdg_bf, bdg);
  build_wcat<<<(4 * DEC * KCAT + 255) / 256, 256, 0, stream>>>(W_ih, W_hh, wcat_bf);
  build_bcat<<<(4 * DEC + 255) / 256, 256, 0, stream>>>(b_ih, b_hh, bcat);
  mean_enc_k<<<(BB * ENC + 127) / 128, 128, 0, stream>>>(enc, meanv, mean_bf);
  // h0, c0: [64,512] = mean[64,128] x W_init^T
  gemm(mean_bf, ENC, winh_bf, ENC, b_inh, h, DEC, h_bf, DEC, BB, DEC, ENC, 0, nullptr, 0);
  gemm(mean_bf, ENC, winc_bf, ENC, b_inc, c, DEC, nullptr, 0, BB, DEC, ENC, 0, nullptr, 0);
  // enc_att (loop-invariant): [B*V,256] = enc[B*V,128] x W_enc^T, stored bf16 in L2
  gemm(enc_bf, ENC, wenc_bf, ENC, b_enc, nullptr, 0, encatt_bf, ATT,
       BB * VV, ATT, ENC, 0, nullptr, 0);
  lens_out_k<<<1, 64, 0, stream>>>(lens, outL);

  // ---- sequential scan: 100 dependent steps, 5 kernels each ----
  for (int t = 0; t < TT; ++t) {
    // [dec_att | sigmoid(beta gate)] = h @ [W_dec;W_beta]^T   -> dg [64,384]
    gemm(h_bf, DEC, wdg_bf, DEC, bdg, dg, ATT + ENC, nullptr, 0,
         BB, ATT + ENC, DEC, 2, nullptr, 0);
    // fused scores + softmax + alphas-out + awe + x assembly (one block per b)
    attn_mega<<<BB, 1024, 0, stream>>>(encatt_bf, enc_bf, dg, w_full, b_full,
                                       emb_bf, caps, h_bf, outA, lens, x_bf, t);
    // LSTM gates: [64,2048] = x[64,896] x [W_ih|W_hh]^T + (b_ih+b_hh)
    gemm(x_bf, KCAT, wcat_bf, KCAT, bcat, gatesb, 4 * DEC, nullptr, 0,
         BB, 4 * DEC, KCAT, 0, nullptr, 0);
    lstm_update<<<(BB * DEC + 255) / 256, 256, 0, stream>>>(gatesb, h, c, h_bf, lens, t);
    // pred = h @ W_final^T + b_final, masked rows -> 0
    gemm(h_bf, DEC, wfin_bf, DEC, b_fin, outP + (size_t)t * VOCAB, TT * VOCAB,
         nullptr, 0, BB, VOCAB, DEC, 0, lens, t);
  }
}
